// EncoderSSGPool_48275432407777
// MI455X (gfx1250) — matrix operand order repaired and checked by reading, not compiled, not run on hardware
//
#include <hip/hip_runtime.h>
#include <hip/hip_bf16.h>
#include <math.h>

// ---------------- problem constants ----------------
#define BB 256
#define NN 300
#define KK 30
#define DD 512
#define EPSF 1e-10f

typedef __attribute__((ext_vector_type(16))) _Float16 v16h;
typedef __attribute__((ext_vector_type(8)))  float    v8f;
typedef __attribute__((ext_vector_type(2)))  _Float16 f16x2;
typedef __attribute__((ext_vector_type(4)))  _Float16 f16x4;

// ---------------- generic batched WMMA GEMM ----------------
// C[b] = op( A[b] @ B[b] )  with epilogue: (+beta*Cold) (+bias[n]) (*mask[b,m]) (relu)
// A indexed as A[bA*b + m*rsA + k*csA], B as B[bB*b + k*rsB + n*csB]
#define TILE 64
#define KT   32
#define LDT  40   // padded f16 stride (80B) to avoid LDS bank conflicts

#define FLAG_RELU 1
#define FLAG_BETA 2

__launch_bounds__(128)
__global__ void gemm_wmma_f16(const float* __restrict__ A, long long bA, int rsA, int csA,
                              const float* __restrict__ B, long long bB, int rsB, int csB,
                              float* __restrict__ C, long long bC, int ldc,
                              const float* __restrict__ bias,
                              const float* __restrict__ mask, int maskStride,
                              int M, int N, int K, int flags)
{
    __shared__ __align__(16) _Float16 As[TILE * LDT];
    __shared__ __align__(16) _Float16 Bs[TILE * LDT];

    const int tid  = threadIdx.x;
    const int lane = tid & 31;
    const int wave = tid >> 5;
    const int wm   = (wave >> 1) * 32;   // wave tile row offset (2x2 waves of 32x32)
    const int wn   = (wave & 1) * 32;
    const int bm   = blockIdx.y * TILE;
    const int bn   = blockIdx.x * TILE;
    const int bz   = blockIdx.z;
    const long long aOff = (long long)bz * bA;
    const long long bOff = (long long)bz * bB;
    const long long cOff = (long long)bz * bC;

    v8f acc[2][2] = {};

    const int fr = lane & 15;          // m (for A) / n (for B) within fragment
    const int kh = (lane >> 4) * 8;    // A-matrix K half-select per ISA layout
    const int kb = (lane >> 4) * 16;   // unused; B fragments also use kh

    // uniform fast-path eligibility (vectorized float4 loads are alignment-safe)
    const bool alignA = (csA == 1) && ((rsA & 3) == 0) && ((bA & 3) == 0);
    const bool alignB = (csB == 1) && ((rsB & 3) == 0) && ((bB & 3) == 0);
    const bool fullM  = (bm + TILE <= M);
    const bool fullN  = (bn + TILE <= N);

    // precomputed per-thread offsets for the vectorized path (advanced each K-step)
    int aR[4], aK4[4], bKK[4], bN4[4];
    long long aRow[4], bRow[4];
    #pragma unroll
    for (int i = 0; i < 4; ++i) {
        int id = tid + i * 128;          // 0..511 float4 slots
        aR[i]  = id >> 3;                // 64 rows x 8 float4
        aK4[i] = (id & 7) * 4;
        aRow[i] = aOff + (long long)(bm + aR[i]) * rsA + aK4[i];
        bKK[i] = id >> 4;                // 32 k-rows x 16 float4
        bN4[i] = (id & 15) * 4;
        bRow[i] = bOff + (long long)bKK[i] * rsB + (bn + bN4[i]);
    }

    for (int k0 = 0; k0 < K; k0 += KT) {
        const bool fullK = (k0 + KT <= K);
        const bool fastA = alignA && fullM && fullK;
        const bool fastB = alignB && fullN && fullK;
        const bool moreK = (k0 + KT < K);

        // ---- stage A tile (64 x 32) f32 -> f16, As[r][k] ----
        if (fastA) {
            float4 va[4];
            #pragma unroll
            for (int i = 0; i < 4; ++i) va[i] = *(const float4*)(A + aRow[i]);
            if (moreK) {   // pull next tile toward cache while this one computes
                #pragma unroll
                for (int i = 0; i < 4; ++i) __builtin_prefetch(A + aRow[i] + KT, 0, 3);
            }
            #pragma unroll
            for (int i = 0; i < 4; ++i) {
                f16x4 h;
                h[0] = (_Float16)va[i].x; h[1] = (_Float16)va[i].y;
                h[2] = (_Float16)va[i].z; h[3] = (_Float16)va[i].w;
                *(f16x4*)&As[aR[i] * LDT + aK4[i]] = h;
            }
        } else {
            // clamp address, load unconditionally, scale by arithmetic keep-factor
            // (multiply keeps the load un-predicable -> all 16 issue, one wait)
            float va[16];
            #pragma unroll
            for (int i = 0; i < 16; ++i) {
                int idx = tid + i * 128;
                int r = idx >> 5, kk = idx & 31;
                int gm = bm + r, gk = k0 + kk;
                int cm = (gm < M) ? gm : (M - 1);
                int ck = (gk < K) ? gk : (K - 1);
                float keep = (gm < M && gk < K) ? 1.0f : 0.0f;
                float v = A[aOff + (long long)cm * rsA + (long long)ck * csA];
                va[i] = v * keep;
            }
            #pragma unroll
            for (int i = 0; i < 16; ++i) {
                int idx = tid + i * 128;
                As[(idx >> 5) * LDT + (idx & 31)] = (_Float16)va[i];
            }
        }

        // ---- stage B tile transposed: Bs[n][k] (64 n x 32 k) ----
        if (fastB) {
            float4 vb[4];
            #pragma unroll
            for (int i = 0; i < 4; ++i) vb[i] = *(const float4*)(B + bRow[i]);
            if (moreK) {
                #pragma unroll
                for (int i = 0; i < 4; ++i)
                    __builtin_prefetch(B + bRow[i] + (long long)KT * rsB, 0, 3);
            }
            #pragma unroll
            for (int i = 0; i < 4; ++i) {
                _Float16 h0 = (_Float16)vb[i].x, h1 = (_Float16)vb[i].y;
                _Float16 h2 = (_Float16)vb[i].z, h3 = (_Float16)vb[i].w;
                _Float16* bp = &Bs[bN4[i] * LDT + bKK[i]];
                bp[0 * LDT] = h0; bp[1 * LDT] = h1; bp[2 * LDT] = h2; bp[3 * LDT] = h3;
            }
        } else {
            float vb[16];
            #pragma unroll
            for (int i = 0; i < 16; ++i) {
                int idx = tid + i * 128;
                int kk = idx >> 6, n = idx & 63;
                int gk = k0 + kk, gn = bn + n;
                int ck = (gk < K) ? gk : (K - 1);
                int cn = (gn < N) ? gn : (N - 1);
                float keep = (gk < K && gn < N) ? 1.0f : 0.0f;
                float v = B[bOff + (long long)ck * rsB + (long long)cn * csB];
                vb[i] = v * keep;
            }
            #pragma unroll
            for (int i = 0; i < 16; ++i) {
                int idx = tid + i * 128;
                Bs[(idx & 63) * LDT + (idx >> 6)] = (_Float16)vb[i];
            }
        }
        __syncthreads();

        // ---- build fragments per ISA 7.12.2 VGPR layouts ----
        v16h af[2], bf[2];
        #pragma unroll
        for (int mi = 0; mi < 2; ++mi) {
            const _Float16* ap = &As[(wm + mi * 16 + fr) * LDT];
            #pragma unroll
            for (int v = 0; v < 8; ++v) {
                int kq = ((v >> 2) * 16) + ((v & 3) * 2) + kh;   // A: K interleave
                f16x2 p = *(const f16x2*)(ap + kq);
                af[mi][2 * v]     = p[0];
                af[mi][2 * v + 1] = p[1];
            }
        }
        #pragma unroll
        for (int ni = 0; ni < 2; ++ni) {
            const _Float16* bp = &Bs[(wn + ni * 16 + fr) * LDT];
            #pragma unroll
            for (int v = 0; v < 8; ++v) {
                int kq = ((v >> 2) * 16) + ((v & 3) * 2) + kh;   // B: K interleave (same as A)
                f16x2 p = *(const f16x2*)(bp + kq);
                bf[ni][2 * v]     = p[0];
                bf[ni][2 * v + 1] = p[1];
            }
        }
        #pragma unroll
        for (int mi = 0; mi < 2; ++mi)
            #pragma unroll
            for (int ni = 0; ni < 2; ++ni)
                acc[mi][ni] = __builtin_amdgcn_wmma_f32_16x16x32_f16(
                    false, af[mi], false, bf[ni], (short)0, acc[mi][ni], false, false);
        __syncthreads();

        // advance vectorized-path pointers
        #pragma unroll
        for (int i = 0; i < 4; ++i) {
            aRow[i] += KT;
            bRow[i] += (long long)KT * rsB;
        }
    }

    // ---- epilogue: C layout = 8 f32/lane, M split at lane 16 ----
    const int mloc = (lane >> 4) * 8;
    const int nloc = lane & 15;
    #pragma unroll
    for (int mi = 0; mi < 2; ++mi) {
        #pragma unroll
        for (int ni = 0; ni < 2; ++ni) {
            int ng = bn + wn + ni * 16 + nloc;
            #pragma unroll
            for (int r = 0; r < 8; ++r) {
                int mg = bm + wm + mi * 16 + mloc + r;
                if (mg < M && ng < N) {
                    long long ci = cOff + (long long)mg * ldc + ng;
                    float v = acc[mi][ni][r];
                    if (flags & FLAG_BETA) v += C[ci];
                    if (bias)              v += bias[ng];
                    if (mask)              v *= mask[(long long)bz * maskStride + mg];
                    if (flags & FLAG_RELU) v = fmaxf(v, 0.0f);
                    C[ci] = v;
                }
            }
        }
    }
}

// ---------------- elementwise / reduction kernels ----------------
__global__ void k_mask(const int* __restrict__ lengths, float* __restrict__ mb) {
    int i = blockIdx.x * 256 + threadIdx.x;
    if (i >= BB * NN) return;
    mb[i] = ((i % NN) < lengths[i / NN]) ? 1.0f : 0.0f;
}

__global__ void k_symadj(const float* __restrict__ araw, float* __restrict__ asym) {
    long long i = (long long)blockIdx.x * 256 + threadIdx.x;
    if (i >= (long long)BB * NN * NN) return;
    long long b = i / (NN * NN);
    int r = (int)((i / NN) % NN), c = (int)(i % NN);
    float a = araw[i] + araw[b * NN * NN + (long long)c * NN + r];
    asym[i] = (a > 0.0f) ? 1.0f : 0.0f;
}

__global__ void k_degdis(const float* __restrict__ asym, float* __restrict__ deg,
                         float* __restrict__ dis) {
    int t = blockIdx.x * 256 + threadIdx.x;
    if (t >= BB * NN) return;
    int b = t / NN, i = t % NN;
    const float* row = asym + (long long)b * NN * NN + (long long)i * NN;
    float s = 0.0f;
    for (int j = 0; j < NN; ++j) s += row[j];
    deg[t] = s;
    float g = s - row[i] + 1.0f;                 // diag set to 1 in dense_gcn copy
    dis[t] = rsqrtf(fmaxf(g, 1.0f));
}

__global__ void k_lap_adjn(const float* __restrict__ asym, const float* __restrict__ deg,
                           const float* __restrict__ dis, float* __restrict__ lap,
                           float* __restrict__ adjn) {
    long long i = (long long)blockIdx.x * 256 + threadIdx.x;
    if (i >= (long long)BB * NN * NN) return;
    int b = (int)(i / (NN * NN));
    int r = (int)((i / NN) % NN), c = (int)(i % NN);
    float a = asym[i];
    lap[i]  = ((r == c) ? deg[b * NN + r] : 0.0f) - a;
    float ag = (r == c) ? 1.0f : a;
    adjn[i] = dis[b * NN + r] * ag * dis[b * NN + c];
}

__global__ void k_gather(const float* __restrict__ emb, const int* __restrict__ nodes,
                         float* __restrict__ x) {
    long long i = (long long)blockIdx.x * 256 + threadIdx.x;
    if (i >= (long long)BB * NN * DD) return;
    long long row = i >> 9;          // / DD
    int d = (int)(i & (DD - 1));
    x[i] = emb[(long long)nodes[row] * DD + d];
}

__global__ void k_softmax(const float* __restrict__ s, const float* __restrict__ mb,
                          float* __restrict__ soft, float* __restrict__ hard) {
    int t = blockIdx.x * 256 + threadIdx.x;
    if (t >= BB * NN) return;
    const float* sp = s + (long long)t * KK;
    float mv = mb[t];
    float mx = sp[0];
    for (int k = 1; k < KK; ++k) mx = fmaxf(mx, sp[k]);
    float e[KK], sum = 0.0f;
    for (int k = 0; k < KK; ++k) { e[k] = __expf(sp[k] - mx); sum += e[k]; }
    float inv = 1.0f / sum;
    int am = 0; float best = -1.0f;
    for (int k = 0; k < KK; ++k) {
        float v = e[k] * inv * mv;
        soft[(long long)t * KK + k] = v;
        if (v > best) { best = v; am = k; }
    }
    for (int k = 0; k < KK; ++k) {
        float sv = soft[(long long)t * KK + k];
        float h  = ((k == am) ? 1.0f : 0.0f) * mv;
        hard[(long long)t * KK + k] = (h - sv) + sv;   // straight-through forward value
    }
}

__global__ void k_colsum(const float* __restrict__ hard, const float* __restrict__ soft,
                         float* __restrict__ csh, float* __restrict__ css) {
    int t = blockIdx.x * 256 + threadIdx.x;
    if (t >= BB * KK) return;
    int b = t / KK, k = t % KK;
    float sh = 0.0f, ss = 0.0f;
    for (int n = 0; n < NN; ++n) {
        long long idx = ((long long)b * NN + n) * KK + k;
        sh += hard[idx]; ss += soft[idx];
    }
    csh[t] = sh; css[t] = ss;
}

__global__ void k_pmat(const float* __restrict__ hard, const float* __restrict__ soft,
                       const float* __restrict__ csh, const float* __restrict__ css,
                       float* __restrict__ P, float* __restrict__ Ps) {
    long long i = (long long)blockIdx.x * 256 + threadIdx.x;
    if (i >= (long long)BB * NN * KK) return;
    int b = (int)(i / (NN * KK));
    int k = (int)(i % KK);
    P[i]  = hard[i] / (csh[b * KK + k] + EPSF);
    Ps[i] = soft[i] / (css[b * KK + k] + EPSF);
}

__global__ void k_dispool(const float* __restrict__ adjp, float* __restrict__ disp) {
    int t = blockIdx.x * 256 + threadIdx.x;
    if (t >= BB * KK) return;
    int b = t / KK, i = t % KK;
    const float* row = adjp + (long long)b * KK * KK + (long long)i * KK;
    float s = 0.0f;
    for (int j = 0; j < KK; ++j) s += (j == i) ? 1.0f : row[j];
    disp[t] = rsqrtf(fmaxf(s, 1.0f));
}

__global__ void k_adjpn(const float* __restrict__ adjp, const float* __restrict__ disp,
                        float* __restrict__ adjpn) {
    int i = blockIdx.x * 256 + threadIdx.x;
    if (i >= BB * KK * KK) return;
    int b = i / (KK * KK);
    int r = (i / KK) % KK, c = i % KK;
    float a = (r == c) ? 1.0f : adjp[i];
    adjpn[i] = disp[b * KK + r] * a * disp[b * KK + c];
}

__global__ void k_featnorm(const float* __restrict__ x, float* __restrict__ out) {
    int b = blockIdx.x, d = threadIdx.x;   // 512 threads
    const float* xp = x + (long long)b * KK * DD;
    float s = 0.0f;
    for (int k = 0; k < KK; ++k) s += xp[k * DD + d];
    s *= (1.0f / KK);
    __shared__ float red[DD];
    red[d] = s * s;
    __syncthreads();
    for (int st = DD / 2; st > 0; st >>= 1) {
        if (d < st) red[d] += red[d + st];
        __syncthreads();
    }
    __shared__ float nrm;
    if (d == 0) nrm = sqrtf(red[0]) + EPSF;
    __syncthreads();
    out[(long long)b * DD + d] = s / nrm;
}

// per-batch sqrt( sum_{ij} ((X - Q) * m_i * m_j)^2 )
__global__ void k_sqdiff(const float* __restrict__ X, const float* __restrict__ Q,
                         const float* __restrict__ mb, float* __restrict__ outb) {
    int b = blockIdx.x, tid = threadIdx.x;
    const float* mrow = mb + (long long)b * NN;
    long long off = (long long)b * NN * NN;
    float s = 0.0f;
    for (int idx = tid; idx < NN * NN; idx += 256) {
        int i = idx / NN, j = idx - i * NN;
        float d = (X[off + idx] - Q[off + idx]) * mrow[i] * mrow[j];
        s += d * d;
    }
    __shared__ float red[256];
    red[tid] = s;
    __syncthreads();
    for (int st = 128; st > 0; st >>= 1) {
        if (tid < st) red[tid] += red[tid + st];
        __syncthreads();
    }
    if (tid == 0) outb[b] = sqrtf(red[0]);
}

__global__ void k_mean256(const float* __restrict__ v, float* __restrict__ out) {
    __shared__ float red[256];
    int t = threadIdx.x;
    red[t] = v[t];
    __syncthreads();
    for (int st = 128; st > 0; st >>= 1) {
        if (t < st) red[t] += red[t + st];
        __syncthreads();
    }
    if (t == 0) out[0] = red[0] * (1.0f / BB);
}

// ---------------- host-side orchestration ----------------
static inline void gemm(hipStream_t st,
                        const float* A, long long bA, int rsA, int csA,
                        const float* B, long long bB, int rsB, int csB,
                        float* C, long long bC, int ldc,
                        const float* bias, const float* mask, int maskStride,
                        int M, int N, int K, int batch, int flags)
{
    dim3 g((N + TILE - 1) / TILE, (M + TILE - 1) / TILE, batch);
    gemm_wmma_f16<<<g, dim3(128), 0, st>>>(A, bA, rsA, csA, B, bB, rsB, csB,
                                           C, bC, ldc, bias, mask, maskStride,
                                           M, N, K, flags);
}

static inline int blks(long long n) { return (int)((n + 255) / 256); }

extern "C" void kernel_launch(void* const* d_in, const int* in_sizes, int n_in,
                              void* d_out, int out_size, void* d_ws, size_t ws_size,
                              hipStream_t stream) {
    (void)in_sizes; (void)n_in; (void)out_size; (void)ws_size;
    const int*   nodes   = (const int*)d_in[0];
    const float* adj_raw = (const float*)d_in[1];
    const int*   lengths = (const int*)d_in[2];
    const float* emb     = (const float*)d_in[3];
    const float* pb1_W1 = (const float*)d_in[4],  *pb1_b1 = (const float*)d_in[5];
    const float* pb1_W2 = (const float*)d_in[6],  *pb1_b2 = (const float*)d_in[7];
    const float* pb1_lW = (const float*)d_in[8],  *pb1_lb = (const float*)d_in[9];
    const float* eb1_W1 = (const float*)d_in[10], *eb1_b1 = (const float*)d_in[11];
    const float* eb1_W2 = (const float*)d_in[12], *eb1_b2 = (const float*)d_in[13];
    const float* eb1_lW = (const float*)d_in[14], *eb1_lb = (const float*)d_in[15];
    const float* ef_W1  = (const float*)d_in[16], *ef_b1  = (const float*)d_in[17];
    const float* ef_W2  = (const float*)d_in[18], *ef_b2  = (const float*)d_in[19];
    const float* ef_lW  = (const float*)d_in[20], *ef_lb  = (const float*)d_in[21];
    float* out = (float*)d_out;

    // ---- workspace bump allocation (floats), 256B aligned ----
    float* w = (float*)d_ws;
    auto alloc = [&](long long n) { float* p = w; w += (n + 63) & ~63LL; return p; };
    const long long SBNN = (long long)BB * NN * NN;   // 23.04M
    const long long SBND = (long long)BB * NN * DD;   // 39.32M
    float* adj_sym = alloc(SBNN);
    float* adj_n   = alloc(SBNN);        // reused as Q scratch after eb1
    float* lap     = alloc(SBNN);
    float* x0      = alloc(SBND);        // reused as xe (eb1 output)
    float* t       = alloc(SBND);
    float* x1      = alloc(SBND);
    float* x2      = alloc(SBND);
    float* s       = alloc((long long)BB * NN * KK);
    float* soft    = alloc((long long)BB * NN * KK);
    float* hard    = alloc((long long)BB * NN * KK);
    float* deg     = alloc(BB * NN);
    float* dis     = alloc(BB * NN);
    float* maskb   = alloc(BB * NN);
    float* csh     = alloc(BB * KK);
    float* css     = alloc(BB * KK);
    float* P       = alloc((long long)BB * NN * KK);
    float* Ps      = alloc((long long)BB * NN * KK);
    float* x_next  = alloc((long long)BB * KK * DD);
    float* tmp_ks  = alloc((long long)BB * KK * NN);
    float* adj_p   = alloc(BB * KK * KK);
    float* adjp_n  = alloc(BB * KK * KK);
    float* disp    = alloc(BB * KK);
    float* L_next  = alloc(BB * KK * KK);
    float* L_soft  = alloc(BB * KK * KK);
    float* PL      = alloc((long long)BB * NN * KK);
    float* xef     = alloc((long long)BB * KK * DD);
    float* sphb    = alloc(BB);
    float* specb   = alloc(BB);
    float* coab    = alloc(BB);
    float* xe = x0;       // alias: x0 dead after eb1's first GEMM
    float* Q  = adj_n;    // alias: adj_n dead after eb1's SpMMs

    const int BN = BB * NN;               // 76800 rows for flattened GEMMs
    const long long bnn = (long long)NN * NN, bnd = (long long)NN * DD;
    const long long bnk = (long long)NN * KK;

    // ---- preprocessing ----
    k_mask  <<<blks(BN), 256, 0, stream>>>(lengths, maskb);
    k_symadj<<<blks(SBNN), 256, 0, stream>>>(adj_raw, adj_sym);
    k_degdis<<<blks(BN), 256, 0, stream>>>(adj_sym, deg, dis);
    k_lap_adjn<<<blks(SBNN), 256, 0, stream>>>(adj_sym, deg, dis, lap, adj_n);
    k_gather<<<blks(SBND), 256, 0, stream>>>(emb, nodes, x0);

    // ---- pb1 block: s = gnn_block(x0, adj, mask, pb1) ----
    gemm(stream, x0, 0, DD, 1, pb1_W1, 0, DD, 1, t, 0, DD, nullptr, nullptr, 0,
         BN, DD, DD, 1, 0);
    gemm(stream, adj_n, bnn, NN, 1, t, bnd, DD, 1, x1, bnd, DD, pb1_b1, maskb, NN,
         NN, DD, NN, BB, FLAG_RELU);
    gemm(stream, x1, 0, DD, 1, pb1_W2, 0, DD, 1, t, 0, DD, nullptr, nullptr, 0,
         BN, DD, DD, 1, 0);
    gemm(stream, adj_n, bnn, NN, 1, t, bnd, DD, 1, x2, bnd, DD, pb1_b2, maskb, NN,
         NN, DD, NN, BB, FLAG_RELU);
    gemm(stream, x1, 0, DD, 1, pb1_lW, 0, KK, 1, s, 0, KK, pb1_lb, nullptr, 0,
         BN, KK, DD, 1, 0);
    gemm(stream, x2, 0, DD, 1, pb1_lW + (long long)DD * KK, 0, KK, 1, s, 0, KK,
         nullptr, nullptr, 0, BN, KK, DD, 1, FLAG_BETA);

    // ---- eb1 block: xe = relu(gnn_block(x0, adj, mask, eb1)) ----
    gemm(stream, x0, 0, DD, 1, eb1_W1, 0, DD, 1, t, 0, DD, nullptr, nullptr, 0,
         BN, DD, DD, 1, 0);
    gemm(stream, adj_n, bnn, NN, 1, t, bnd, DD, 1, x1, bnd, DD, eb1_b1, maskb, NN,
         NN, DD, NN, BB, FLAG_RELU);
    gemm(stream, x1, 0, DD, 1, eb1_W2, 0, DD, 1, t, 0, DD, nullptr, nullptr, 0,
         BN, DD, DD, 1, 0);
    gemm(stream, adj_n, bnn, NN, 1, t, bnd, DD, 1, x2, bnd, DD, eb1_b2, maskb, NN,
         NN, DD, NN, BB, FLAG_RELU);
    gemm(stream, x1, 0, DD, 1, eb1_lW, 0, DD, 1, xe, 0, DD, eb1_lb, nullptr, 0,
         BN, DD, DD, 1, 0);
    gemm(stream, x2, 0, DD, 1, eb1_lW + (long long)DD * DD, 0, DD, 1, xe, 0, DD,
         nullptr, nullptr, 0, BN, DD, DD, 1, FLAG_BETA | FLAG_RELU);

    // ---- ssgpool ----
    k_softmax<<<blks(BN), 256, 0, stream>>>(s, maskb, soft, hard);
    // x_next = s_hard^T @ xe
    gemm(stream, hard, bnk, 1, KK, xe, bnd, DD, 1, x_next, (long long)KK * DD, DD,
         nullptr, nullptr, 0, KK, DD, NN, BB, 0);
    // adj_p = s^T @ adj @ s
    gemm(stream, hard, bnk, 1, KK, adj_sym, bnn, NN, 1, tmp_ks, (long long)KK * NN, NN,
         nullptr, nullptr, 0, KK, NN, NN, BB, 0);
    gemm(stream, tmp_ks, (long long)KK * NN, NN, 1, hard, bnk, KK, 1, adj_p,
         (long long)KK * KK, KK, nullptr, nullptr, 0, KK, KK, NN, BB, 0);
    // L_next = s^T @ L @ s  (hard)
    gemm(stream, hard, bnk, 1, KK, lap, bnn, NN, 1, tmp_ks, (long long)KK * NN, NN,
         nullptr, nullptr, 0, KK, NN, NN, BB, 0);
    gemm(stream, tmp_ks, (long long)KK * NN, NN, 1, hard, bnk, KK, 1, L_next,
         (long long)KK * KK, KK, nullptr, nullptr, 0, KK, KK, NN, BB, 0);
    // L_next_soft = s_soft^T @ L @ s_soft
    gemm(stream, soft, bnk, 1, KK, lap, bnn, NN, 1, tmp_ks, (long long)KK * NN, NN,
         nullptr, nullptr, 0, KK, NN, NN, BB, 0);
    gemm(stream, tmp_ks, (long long)KK * NN, NN, 1, soft, bnk, KK, 1, L_soft,
         (long long)KK * KK, KK, nullptr, nullptr, 0, KK, KK, NN, BB, 0);
    // P, P_soft
    k_colsum<<<blks(BB * KK), 256, 0, stream>>>(hard, soft, csh, css);
    k_pmat  <<<blks((long long)BB * NN * KK), 256, 0, stream>>>(hard, soft, csh, css, P, Ps);

    // ---- ef block on pooled graph (mask=None) ----
    k_dispool<<<blks(BB * KK), 256, 0, stream>>>(adj_p, disp);
    k_adjpn  <<<blks(BB * KK * KK), 256, 0, stream>>>(adj_p, disp, adjp_n);
    const int BK = BB * KK;               // 7680 rows
    gemm(stream, x_next, 0, DD, 1, ef_W1, 0, DD, 1, t, 0, DD, nullptr, nullptr, 0,
         BK, DD, DD, 1, 0);
    gemm(stream, adjp_n, (long long)KK * KK, KK, 1, t, (long long)KK * DD, DD, 1,
         x1, (long long)KK * DD, DD, ef_b1, nullptr, 0, KK, DD, KK, BB, FLAG_RELU);
    gemm(stream, x1, 0, DD, 1, ef_W2, 0, DD, 1, t, 0, DD, nullptr, nullptr, 0,
         BK, DD, DD, 1, 0);
    gemm(stream, adjp_n, (long long)KK * KK, KK, 1, t, (long long)KK * DD, DD, 1,
         x2, (long long)KK * DD, DD, ef_b2, nullptr, 0, KK, DD, KK, BB, FLAG_RELU);
    gemm(stream, x1, 0, DD, 1, ef_lW, 0, DD, 1, xef, 0, DD, ef_lb, nullptr, 0,
         BK, DD, DD, 1, 0);
    gemm(stream, x2, 0, DD, 1, ef_lW + (long long)DD * DD, 0, DD, 1, xef, 0, DD,
         nullptr, nullptr, 0, BK, DD, DD, 1, FLAG_BETA);

    // feature_out = normalize(mean over clusters)
    k_featnorm<<<BB, DD, 0, stream>>>(xef, out);

    // ---- spectral losses ----
    // hard: Q = (P @ L_next) @ P^T ; spec_hard[b] = ||(L - Q)*mm||_F
    gemm(stream, P, bnk, KK, 1, L_next, (long long)KK * KK, KK, 1, PL, bnk, KK,
         nullptr, nullptr, 0, NN, KK, KK, BB, 0);
    gemm(stream, PL, bnk, KK, 1, P, bnk, 1, KK, Q, bnn, NN,
         nullptr, nullptr, 0, NN, NN, KK, BB, 0);
    k_sqdiff<<<BB, 256, 0, stream>>>(lap, Q, maskb, sphb);
    // soft
    gemm(stream, Ps, bnk, KK, 1, L_soft, (long long)KK * KK, KK, 1, PL, bnk, KK,
         nullptr, nullptr, 0, NN, KK, KK, BB, 0);
    gemm(stream, PL, bnk, KK, 1, Ps, bnk, 1, KK, Q, bnn, NN,
         nullptr, nullptr, 0, NN, NN, KK, BB, 0);
    k_sqdiff<<<BB, 256, 0, stream>>>(lap, Q, maskb, specb);
    // coarsen: Q = s_hard @ s_hard^T ; coarsen[b] = ||(ori_adj - Q)*mm||_F
    gemm(stream, hard, bnk, KK, 1, hard, bnk, 1, KK, Q, bnn, NN,
         nullptr, nullptr, 0, NN, NN, KK, BB, 0);
    k_sqdiff<<<BB, 256, 0, stream>>>(adj_sym, Q, maskb, coab);

    // ---- scalar means: (feature_out, spec, spec_hard, coarsen) ----
    k_mean256<<<1, 256, 0, stream>>>(specb, out + (long long)BB * DD + 0);
    k_mean256<<<1, 256, 0, stream>>>(sphb,  out + (long long)BB * DD + 1);
    k_mean256<<<1, 256, 0, stream>>>(coab,  out + (long long)BB * DD + 2);
}
